// LinearAttention_72911364816914
// MI455X (gfx1250) — matrix operand order repaired and checked by reading, not compiled, not run on hardware
//
#include <hip/hip_runtime.h>
#include <hip/hip_bf16.h>

typedef __bf16 bf16;
typedef __attribute__((ext_vector_type(16))) __bf16 v16bf;
typedef __attribute__((ext_vector_type(8)))  __bf16 v8bf;
typedef __attribute__((ext_vector_type(8)))  float  v8f;

#define HEADS 8
#define DIMH  64
#define BATCH 16
#define CH    512
#define NN    4096   // 64*64

// round-to-nearest-even f32 -> bf16
__device__ __forceinline__ bf16 f2bf(float f) {
    union { float f; unsigned u; } v; v.f = f;
    unsigned r = (v.u + 0x7FFFu + ((v.u >> 16) & 1u)) >> 16;
    unsigned short s = (unsigned short)r;
    return __builtin_bit_cast(bf16, s);
}

// CDNA5 async DMA: global -> LDS, 16 bytes per lane. LDS address = low 32 bits
// of the flat pointer (ISA 10.2: LDS aperture truncates to addr[31:0]).
__device__ __forceinline__ void async_load_b128(unsigned lds_off, const void* gptr) {
    asm volatile("global_load_async_to_lds_b128 %0, %1, off"
                 :: "v"(lds_off), "v"(gptr) : "memory");
}
__device__ __forceinline__ void wait_async0() {
    asm volatile("s_wait_asynccnt 0" ::: "memory");
}

// A fragment, 16x32 bf16, A row-major with leading dim lda (elements).
// Lane M = lane&15; lanes 0-15 hold K 0..15, lanes 16-31 hold K 16..31.
__device__ __forceinline__ v16bf load_a_frag(const bf16* __restrict__ A, int lda, int lane) {
    int M = lane & 15, half = lane >> 4;
    const bf16* p = A + (size_t)M * lda + half * 16;
    return *(const v16bf*)p;
}

// B fragment, 32(K)x16(N), from an N-major array BT[n][k], ld = ldb elements.
__device__ __forceinline__ v16bf load_b_frag(const bf16* __restrict__ BT, int ldb, int lane) {
    int N = lane & 15, kk = (lane >> 4) * 16;
    return *(const v16bf*)(BT + (size_t)N * ldb + kk);
}

// ---------------- K0a: x (b,512,4096) f32  ->  xbT (b,4096,512) bf16 ----------------
__global__ __launch_bounds__(256) void transpose_convert_x(const float* __restrict__ x,
                                                           bf16* __restrict__ xbT) {
    int b = blockIdx.z, n0 = blockIdx.x * 64, c0 = blockIdx.y * 64;
    __shared__ float tile[64][65];
    int t = threadIdx.x;
    int col = t & 63, rb = t >> 6;
    const float* src = x + ((size_t)b * CH + c0) * NN + n0;
    #pragma unroll
    for (int j = 0; j < 16; ++j) {
        int r = rb + j * 4;
        tile[r][col] = src[(size_t)r * NN + col];
    }
    __syncthreads();
    bf16* dst = xbT + ((size_t)b * NN + n0) * CH + c0;
    #pragma unroll
    for (int j = 0; j < 16; ++j) {
        int nrow = rb + j * 4;
        dst[(size_t)nrow * CH + col] = f2bf(tile[col][nrow]);
    }
}

// ---------------- K0b/c: plain f32 -> bf16 convert ----------------
__global__ __launch_bounds__(256) void convert_f32_bf16(const float* __restrict__ src,
                                                        bf16* __restrict__ dst, int n) {
    int i = blockIdx.x * 256 + threadIdx.x;
    if (i < n) dst[i] = f2bf(src[i]);
}

// ---------------- K1: qkv GEMM. M=1536 K=512 N=4096 per batch ----------------
// Block tile 128x128, 8 waves (4M x 2N). B tile (128n x 32k bf16, 8 KB) staged in LDS by
// the CDNA5 async DMA path, double-buffered; A fragments direct from global (L2-resident).
__global__ __launch_bounds__(256) void gemm_qkv_kernel(const bf16* __restrict__ Wq,
                                                       const bf16* __restrict__ xbT,
                                                       bf16*  __restrict__ qT,
                                                       float* __restrict__ kf,
                                                       bf16*  __restrict__ vb) {
    const int b  = blockIdx.z;
    const int m0 = blockIdx.y * 128;
    const int n0 = blockIdx.x * 128;
    const int tid = threadIdx.x, lane = tid & 31, wid = tid >> 5;
    const int wm = (wid & 3) * 32, wn = (wid >> 2) * 64;
    const bf16* Bbase = xbT + (size_t)b * NN * CH;

    __shared__ __align__(16) bf16 bsh[2][128 * 32];   // 2 x 8 KB

    const int lrow  = tid >> 2;        // 0..63
    const int lkoff = (tid & 3) * 8;   // 0,8,16,24 (bf16 elems)

    #define ISSUE_B(bufi, kk0)                                                        \
        do {                                                                          \
            async_load_b128((unsigned)(uintptr_t)&bsh[bufi][lrow * 32 + lkoff],       \
                            Bbase + (size_t)(n0 + lrow) * CH + (kk0) + lkoff);        \
            async_load_b128((unsigned)(uintptr_t)&bsh[bufi][(64 + lrow) * 32 + lkoff],\
                            Bbase + (size_t)(n0 + 64 + lrow) * CH + (kk0) + lkoff);   \
        } while (0)

    v8f acc[2][4] = {};
    ISSUE_B(0, 0);
    int buf = 0;
    for (int k0 = 0; k0 < CH; k0 += 32) {
        wait_async0();          // this wave's async writes to LDS done
        __syncthreads();        // all waves' writes visible; prev buffer free for reuse
        if (k0 + 32 < CH) ISSUE_B(buf ^ 1, k0 + 32);

        v16bf a[2], bb[4];
        #pragma unroll
        for (int i = 0; i < 2; ++i)
            a[i] = load_a_frag(Wq + (size_t)(m0 + wm + i * 16) * CH + k0, CH, lane);
        #pragma unroll
        for (int j = 0; j < 4; ++j)
            bb[j] = *(const v16bf*)(&bsh[buf][(wn + j * 16 + (lane & 15)) * 32 + (lane >> 4) * 16]);
        if (k0 + 32 < CH)
            __builtin_prefetch(Wq + (size_t)(m0 + wm + (lane & 15)) * CH + k0 + 32, 0, 1);
        #pragma unroll
        for (int i = 0; i < 2; ++i)
            #pragma unroll
            for (int j = 0; j < 4; ++j)
                acc[i][j] = __builtin_amdgcn_wmma_f32_16x16x32_bf16(
                    false, a[i], false, bb[j], (short)0, acc[i][j], false, false);
        buf ^= 1;
    }
    #undef ISSUE_B

    #pragma unroll
    for (int i = 0; i < 2; ++i)
        #pragma unroll
        for (int j = 0; j < 4; ++j)
            #pragma unroll
            for (int e = 0; e < 8; ++e) {
                int o = m0 + wm + i * 16 + e + ((lane >> 4) << 3);
                int n = n0 + wn + j * 16 + (lane & 15);
                float val = acc[i][j][e];
                int which = o >> 9, r = o & 511;
                if (which == 0) {
                    int h = r >> 6, c = r & 63;
                    qT[(((size_t)b * HEADS + h) * NN + n) * DIMH + c] = f2bf(val);
                } else if (which == 1) {
                    kf[((size_t)b * CH + r) * NN + n] = val;
                } else {
                    vb[((size_t)b * CH + r) * NN + n] = f2bf(val);
                }
            }
}

// ---------------- K2: softmax over n (4096) for each of 16*512 k-rows ----------------
__global__ __launch_bounds__(256) void softmax_rows_kernel(const float* __restrict__ kf,
                                                           bf16* __restrict__ kb) {
    const size_t row = blockIdx.x;
    const float* src = kf + row * NN;
    bf16* dst = kb + row * NN;
    __shared__ float red[256];
    int t = threadIdx.x;
    float m = -3.4e38f;
    for (int i = t; i < NN; i += 256) m = fmaxf(m, src[i]);
    red[t] = m; __syncthreads();
    for (int s = 128; s > 0; s >>= 1) { if (t < s) red[t] = fmaxf(red[t], red[t + s]); __syncthreads(); }
    m = red[0]; __syncthreads();
    float sum = 0.f;
    for (int i = t; i < NN; i += 256) sum += __expf(src[i] - m);
    red[t] = sum; __syncthreads();
    for (int s = 128; s > 0; s >>= 1) { if (t < s) red[t] += red[t + s]; __syncthreads(); }
    float inv = 1.0f / red[0];
    for (int i = t; i < NN; i += 256) dst[i] = f2bf(__expf(src[i] - m) * inv);
}

// ---------------- K3: context[c,d] = sum_n softk[c,n] * v[d,n]  (64x64 per bh) --------
__global__ __launch_bounds__(256) void context_kernel(const bf16* __restrict__ kb,
                                                      const bf16* __restrict__ vb,
                                                      bf16* __restrict__ ctxT) {
    const int bh = blockIdx.x;
    const int b = bh >> 3, h = bh & 7;
    const int tid = threadIdx.x, lane = tid & 31, wid = tid >> 5;
    const bf16* Ab = kb + ((size_t)b * CH + h * DIMH) * NN;
    const bf16* Bb = vb + ((size_t)b * CH + h * DIMH) * NN;
    __shared__ float cacc[DIMH * DIMH];
    for (int i = tid; i < DIMH * DIMH; i += 256) cacc[i] = 0.f;
    __syncthreads();
    v8f acc[4][4] = {};
    const int kbase = wid * 512;
    for (int k0 = kbase; k0 < kbase + 512; k0 += 32) {
        v16bf a[4], bb[4];
        #pragma unroll
        for (int i = 0; i < 4; ++i)
            a[i] = load_a_frag(Ab + (size_t)(i * 16) * NN + k0, NN, lane);
        #pragma unroll
        for (int j = 0; j < 4; ++j)
            bb[j] = load_b_frag(Bb + (size_t)(j * 16) * NN + k0, NN, lane);
        #pragma unroll
        for (int i = 0; i < 4; ++i)
            #pragma unroll
            for (int j = 0; j < 4; ++j)
                acc[i][j] = __builtin_amdgcn_wmma_f32_16x16x32_bf16(
                    false, a[i], false, bb[j], (short)0, acc[i][j], false, false);
    }
    #pragma unroll
    for (int i = 0; i < 4; ++i)
        #pragma unroll
        for (int j = 0; j < 4; ++j)
            #pragma unroll
            for (int e = 0; e < 8; ++e) {
                int c = i * 16 + e + ((lane >> 4) << 3);
                int d = j * 16 + (lane & 15);
                atomicAdd(&cacc[c * DIMH + d], acc[i][j][e]);
            }
    __syncthreads();
    bf16* out = ctxT + (size_t)bh * DIMH * DIMH;
    for (int i = tid; i < DIMH * DIMH; i += 256) {
        int d = i >> 6, c = i & 63;
        out[i] = f2bf(cacc[c * DIMH + d]);     // ctxT[d][c] = context[c][d]
    }
}

// ---------------- K4: out[d,n] = sum_c ctxT[d,c] * q[c,n] ----------------
__global__ __launch_bounds__(256) void attend_out_kernel(const bf16* __restrict__ ctxT,
                                                         const bf16* __restrict__ qT,
                                                         bf16* __restrict__ outT) {
    const int bh = blockIdx.y;
    const int b = bh >> 3, h = bh & 7;
    const int n0 = blockIdx.x * 256;
    const int tid = threadIdx.x, lane = tid & 31, wid = tid >> 5;
    const int wn = wid * 32;
    const bf16* A = ctxT + (size_t)bh * DIMH * DIMH;
    const bf16* B = qT + ((size_t)bh * NN + n0 + wn) * DIMH;
    v8f acc[4][2] = {};
    #pragma unroll
    for (int k0 = 0; k0 < DIMH; k0 += 32) {
        v16bf a[4], bb[2];
        #pragma unroll
        for (int i = 0; i < 4; ++i)
            a[i] = load_a_frag(A + (size_t)(i * 16) * DIMH + k0, DIMH, lane);
        #pragma unroll
        for (int j = 0; j < 2; ++j)
            bb[j] = load_b_frag(B + (size_t)(j * 16) * DIMH + k0, DIMH, lane);
        #pragma unroll
        for (int i = 0; i < 4; ++i)
            #pragma unroll
            for (int j = 0; j < 2; ++j)
                acc[i][j] = __builtin_amdgcn_wmma_f32_16x16x32_bf16(
                    false, a[i], false, bb[j], (short)0, acc[i][j], false, false);
    }
    #pragma unroll
    for (int i = 0; i < 4; ++i)
        #pragma unroll
        for (int j = 0; j < 2; ++j)
            #pragma unroll
            for (int e = 0; e < 8; ++e) {
                int d = i * 16 + e + ((lane >> 4) << 3);
                int n = n0 + wn + j * 16 + (lane & 15);
                outT[((size_t)b * NN + n) * CH + h * DIMH + d] = f2bf(acc[i][j][e]);
            }
}

// ---------------- K5: y = w_out @ out + b_out. M=512 K=512 N=4096 per batch ----------
__global__ __launch_bounds__(256) void gemm_out_kernel(const bf16* __restrict__ Wo,
                                                       const bf16* __restrict__ outT,
                                                       const float* __restrict__ bias,
                                                       float* __restrict__ y) {
    const int b  = blockIdx.z;
    const int m0 = blockIdx.y * 128;
    const int n0 = blockIdx.x * 128;
    const int tid = threadIdx.x, lane = tid & 31, wid = tid >> 5;
    const int wm = (wid & 3) * 32, wn = (wid >> 2) * 64;
    const bf16* Bbase = outT + (size_t)b * NN * CH;

    __shared__ __align__(16) bf16 bsh[2][128 * 32];

    const int lrow  = tid >> 2;
    const int lkoff = (tid & 3) * 8;

    #define ISSUE_B(bufi, kk0)                                                        \
        do {                                                                          \
            async_load_b128((unsigned)(uintptr_t)&bsh[bufi][lrow * 32 + lkoff],       \
                            Bbase + (size_t)(n0 + lrow) * CH + (kk0) + lkoff);        \
            async_load_b128((unsigned)(uintptr_t)&bsh[bufi][(64 + lrow) * 32 + lkoff],\
                            Bbase + (size_t)(n0 + 64 + lrow) * CH + (kk0) + lkoff);   \
        } while (0)

    v8f acc[2][4] = {};
    ISSUE_B(0, 0);
    int buf = 0;
    for (int k0 = 0; k0 < CH; k0 += 32) {
        wait_async0();
        __syncthreads();
        if (k0 + 32 < CH) ISSUE_B(buf ^ 1, k0 + 32);

        v16bf a[2], bb[4];
        #pragma unroll
        for (int i = 0; i < 2; ++i)
            a[i] = load_a_frag(Wo + (size_t)(m0 + wm + i * 16) * CH + k0, CH, lane);
        #pragma unroll
        for (int j = 0; j < 4; ++j)
            bb[j] = *(const v16bf*)(&bsh[buf][(wn + j * 16 + (lane & 15)) * 32 + (lane >> 4) * 16]);
        #pragma unroll
        for (int i = 0; i < 2; ++i)
            #pragma unroll
            for (int j = 0; j < 4; ++j)
                acc[i][j] = __builtin_amdgcn_wmma_f32_16x16x32_bf16(
                    false, a[i], false, bb[j], (short)0, acc[i][j], false, false);
        buf ^= 1;
    }
    #undef ISSUE_B

    #pragma unroll
    for (int i = 0; i < 2; ++i)
        #pragma unroll
        for (int j = 0; j < 4; ++j)
            #pragma unroll
            for (int e = 0; e < 8; ++e) {
                int o = m0 + wm + i * 16 + e + ((lane >> 4) << 3);
                int n = n0 + wn + j * 16 + (lane & 15);
                y[((size_t)b * CH + o) * NN + n] = acc[i][j][e] + bias[o];
            }
}

extern "C" void kernel_launch(void* const* d_in, const int* in_sizes, int n_in,
                              void* d_out, int out_size, void* d_ws, size_t ws_size,
                              hipStream_t stream) {
    const float* x     = (const float*)d_in[0];   // (16,512,64,64)
    const float* w_qkv = (const float*)d_in[1];   // (1536,512)
    const float* w_out = (const float*)d_in[2];   // (512,512)
    const float* b_out = (const float*)d_in[3];   // (512,)
    float* y = (float*)d_out;

    char* w = (char*)d_ws;
    bf16*  xbT  = (bf16*)w;  w += (size_t)BATCH * NN * CH * 2;
    bf16*  wqb  = (bf16*)w;  w += (size_t)3 * CH * CH * 2;
    bf16*  wob  = (bf16*)w;  w += (size_t)CH * CH * 2;
    bf16*  qT   = (bf16*)w;  w += (size_t)BATCH * HEADS * NN * DIMH * 2;
    float* kf   = (float*)w; w += (size_t)BATCH * CH * NN * 4;
    bf16*  vb   = (bf16*)w;  w += (size_t)BATCH * CH * NN * 2;
    bf16*  kb   = (bf16*)w;  w += (size_t)BATCH * CH * NN * 2;
    bf16*  ctxT = (bf16*)w;  w += (size_t)BATCH * HEADS * DIMH * DIMH * 2;
    bf16*  outT = (bf16*)w;  w += (size_t)BATCH * NN * CH * 2;

    transpose_convert_x<<<dim3(NN / 64, CH / 64, BATCH), 256, 0, stream>>>(x, xbT);
    convert_f32_bf16<<<(3 * CH * CH + 255) / 256, 256, 0, stream>>>(w_qkv, wqb, 3 * CH * CH);
    convert_f32_bf16<<<(CH * CH + 255) / 256, 256, 0, stream>>>(w_out, wob, CH * CH);

    gemm_qkv_kernel<<<dim3(NN / 128, 3 * CH / 128, BATCH), 256, 0, stream>>>(wqb, xbT, qT, kf, vb);
    softmax_rows_kernel<<<BATCH * CH, 256, 0, stream>>>(kf, kb);
    context_kernel<<<BATCH * HEADS, 256, 0, stream>>>(kb, vb, ctxT);
    attend_out_kernel<<<dim3(NN / 256, BATCH * HEADS), 256, 0, stream>>>(ctxT, qT, outT);
    gemm_out_kernel<<<dim3(NN / 128, CH / 128, BATCH), 256, 0, stream>>>(wob, outT, b_out, y);
}
